// torchLSTM_cell_mc_48258252538452
// MI455X (gfx1250) — compile-verified, matrix operand order checked by reading, not run on hardware
//
#include <hip/hip_runtime.h>
#include <hip/hip_bf16.h>

// ---------------------------------------------------------------------------
// LSTM (torchLSTM_cell_mc): NT=365, NGRID=1000, NX=20, HID=256, NY=1
// Strategy: bf16 WMMA (v_wmma_f32_16x16x32_bf16) with fp32 accum/state.
// Fused per-step GEMM: gates = x0_t @ W_ih^T + h @ W_hh^T + (b_ih+b_hh).
// ---------------------------------------------------------------------------

#define NT   365
#define NG   1000
#define NX   20
#define HID  256
#define G4   1024   // 4*HID

typedef __attribute__((ext_vector_type(16))) __bf16          v16bf;
typedef __attribute__((ext_vector_type(8)))  float           v8f;
typedef __attribute__((ext_vector_type(8)))  unsigned short  u16x8;

__device__ __forceinline__ unsigned short f2bf(float f) {
    union { float f; unsigned u; } v; v.f = f;
    unsigned u = v.u;
    if ((u & 0x7fffffffu) > 0x7f800000u) return 0x7fc0;          // NaN
    return (unsigned short)((u + 0x7fffu + ((u >> 16) & 1u)) >> 16); // RNE
}
__device__ __forceinline__ float bf2f(unsigned short h) {
    union { unsigned u; float f; } v; v.u = ((unsigned)h) << 16;
    return v.f;
}
__device__ __forceinline__ float sigm(float x) { return 1.f / (1.f + __expf(-x)); }

// Build a v16bf fragment from two non-adjacent 16-byte segments.
__device__ __forceinline__ v16bf load_frag2(const unsigned short* p0,
                                            const unsigned short* p1) {
    union { u16x8 u[2]; v16bf v; } f;
    f.u[0] = *reinterpret_cast<const u16x8*>(p0);
    f.u[1] = *reinterpret_cast<const u16x8*>(p1);
    return f.v;
}

// ---------------------------------------------------------------------------
// Init: zero h0 and c, fuse biases.
// ---------------------------------------------------------------------------
__global__ void init_state(unsigned short* __restrict__ h0,
                           float* __restrict__ c,
                           const float* __restrict__ b_ih,
                           const float* __restrict__ b_hh,
                           float* __restrict__ bias) {
    int i = blockIdx.x * 256 + threadIdx.x;
    if (i < NG * HID) { h0[i] = 0; c[i] = 0.f; }
    if (i < G4) bias[i] = b_ih[i] + b_hh[i];
}

// ---------------------------------------------------------------------------
// Pack W[g,k] (G4 x HID, row-major fp32) into WMMA B-fragments (bf16):
// frag index = ((nt*8 + kc)*32 + lane)*16 + e  with
//   lane<16 : N = nt*16+lane,   K = kc*32 +  e     (e<8) | kc*32+16+(e-8)
//   lane>=16: N = nt*16+lane-16,K = kc*32+8+e      (e<8) | kc*32+24+(e-8)
// (16-bit B-matrix 32x16 per-lane layout, ISA 7.12.2)
// ---------------------------------------------------------------------------
__global__ void pack_w(const float* __restrict__ W, unsigned short* __restrict__ F) {
    int idx  = blockIdx.x * 256 + threadIdx.x;     // 0 .. G4*HID-1
    int e    = idx & 15;
    int lane = (idx >> 4) & 31;
    int kc   = (idx >> 9) & 7;
    int nt   = idx >> 12;
    int laneN = lane & 15, half = lane >> 4;
    int g = nt * 16 + laneN;
    int k = kc * 32 + half * 8 + e + (e >= 8 ? 8 : 0);
    F[idx] = f2bf(W[g * HID + k]);
}

// ---------------------------------------------------------------------------
// linearIn + ReLU -> bf16: x0[row, h] = relu(sum_k x[row,k]*W_in[h,k] + b_in[h])
// one block per row (365,000 rows), 256 threads = HID.
// ---------------------------------------------------------------------------
__global__ void lin_in(const float* __restrict__ x,
                       const float* __restrict__ W_in,
                       const float* __restrict__ b_in,
                       unsigned short* __restrict__ x0) {
    __shared__ float xs[NX];
    size_t row = blockIdx.x;
    int h = threadIdx.x;
    if (h < NX) xs[h] = x[row * NX + h];
    __syncthreads();
    float s = b_in[h];
#pragma unroll
    for (int k = 0; k < NX; ++k) s += xs[k] * W_in[h * NX + k];
    x0[row * HID + h] = f2bf(fmaxf(s, 0.f));
}

// ---------------------------------------------------------------------------
// One LSTM step. grid = (32 m-tiles, 4 column groups), block = 256 (8 waves).
// wave w: wm = w&1 selects 16-row subtile, wq = w>>1 selects gate q.
// acc[nn] covers N = q*256 + cg*64 + nn*16 .. +15.  K loop: 8 chunks x0 + 8 h.
// ---------------------------------------------------------------------------
__global__ void lstm_step(const unsigned short* __restrict__ x0t,
                          const unsigned short* __restrict__ wihF,
                          const unsigned short* __restrict__ whhF,
                          const float* __restrict__ bias,
                          const unsigned short* __restrict__ h_in,
                          float* __restrict__ c,
                          unsigned short* __restrict__ h_out,
                          unsigned short* __restrict__ hst) {
    __shared__ float gl[4 * 32 * 64];                 // 32 KB gate staging

    const int tid   = threadIdx.x;
    const int wave  = tid >> 5;
    const int lane  = tid & 31;
    const int wm    = wave & 1;                       // M subtile
    const int wq    = wave >> 1;                      // gate index 0..3
    const int mtile = blockIdx.x;                     // 0..31
    const int cg    = blockIdx.y;                     // 0..3 (64 hidden cols)
    const int laneN = lane & 15;
    const int half  = lane >> 4;

    const int m0   = mtile * 32 + wm * 16;
    const int rowA = min(m0 + laneN, NG - 1);         // clamp tail rows

    v8f acc[4] = {v8f{}, v8f{}, v8f{}, v8f{}};

    // ---- phase 1: x0_t @ W_ih^T ; phase 2: h @ W_hh^T --------------------
#pragma unroll 1
    for (int phase = 0; phase < 2; ++phase) {
        const unsigned short* A = phase ? h_in : x0t;
        const unsigned short* B = phase ? whhF : wihF;
#pragma unroll 1
        for (int kc = 0; kc < 8; ++kc) {
            const size_t abase = (size_t)rowA * HID + kc * 32 + half * 8;
            v16bf a = load_frag2(A + abase, A + abase + 16);
#pragma unroll
            for (int nn = 0; nn < 4; ++nn) {
                const int nt = wq * 16 + cg * 4 + nn;
                const unsigned short* bp = B + ((size_t)(nt * 8 + kc) * 32 + lane) * 16;
                v16bf b = load_frag2(bp, bp + 8);
                acc[nn] = __builtin_amdgcn_wmma_f32_16x16x32_bf16(
                    false, a, false, b, (short)0, acc[nn], false, false);
            }
        }
    }

    // ---- bias + stage gates in LDS --------------------------------------
#pragma unroll
    for (int nn = 0; nn < 4; ++nn) {
        const int ncol = wq * 256 + cg * 64 + nn * 16 + laneN; // global gate col
        const float bv = bias[ncol];
        const int col = nn * 16 + laneN;                       // 0..63 in group
#pragma unroll
        for (int j = 0; j < 8; ++j) {
            const int ml = wm * 16 + half * 8 + j;             // 0..31 local row
            gl[(wq * 32 + ml) * 64 + col] = acc[nn][j] + bv;
        }
    }
    __syncthreads();

    // ---- elementwise LSTM cell update (8 contiguous cols per thread) ----
    const int ml  = tid >> 3;                // 0..31
    const int cl0 = (tid & 7) * 8;           // 0..56
    const int m   = mtile * 32 + ml;
    if (m < NG) {
        const size_t base = (size_t)m * HID + cg * 64 + cl0;
#pragma unroll
        for (int j = 0; j < 8; ++j) {
            const int cl = cl0 + j;
            const float iv = gl[(0 * 32 + ml) * 64 + cl];
            const float fv = gl[(1 * 32 + ml) * 64 + cl];
            const float gv = gl[(2 * 32 + ml) * 64 + cl];
            const float ov = gl[(3 * 32 + ml) * 64 + cl];
            const float cOld = c[base + j];
            const float cNew = sigm(fv) * cOld + sigm(iv) * tanhf(gv);
            const float hNew = sigm(ov) * tanhf(cNew);
            c[base + j] = cNew;
            const unsigned short hb = f2bf(hNew);
            h_out[base + j] = hb;
            hst[base + j]   = hb;
        }
    }
}

// ---------------------------------------------------------------------------
// out[row] = sum_h hs[row,h]*W_out[h] + b_out ; one wave per row.
// ---------------------------------------------------------------------------
__global__ void lin_out(const unsigned short* __restrict__ hs,
                        const float* __restrict__ W_out,
                        const float* __restrict__ b_out,
                        float* __restrict__ out) {
    const int wave = threadIdx.x >> 5;
    const int lane = threadIdx.x & 31;
    const size_t row = (size_t)blockIdx.x * 8 + wave;
    if (row >= (size_t)NT * NG) return;               // wave-uniform exit
    const u16x8 hv = *reinterpret_cast<const u16x8*>(hs + row * HID + lane * 8);
    float s = 0.f;
#pragma unroll
    for (int j = 0; j < 8; ++j) s += bf2f(hv[j]) * W_out[lane * 8 + j];
#pragma unroll
    for (int m = 16; m > 0; m >>= 1) s += __shfl_xor(s, m, 32);
    if (lane == 0) out[row] = s + b_out[0];
}

// ---------------------------------------------------------------------------
extern "C" void kernel_launch(void* const* d_in, const int* in_sizes, int n_in,
                              void* d_out, int out_size, void* d_ws, size_t ws_size,
                              hipStream_t stream) {
    const float* x     = (const float*)d_in[0];
    /* d_in[1] = wt_ih (zeros, eval mode: unused) */
    const float* W_in  = (const float*)d_in[2];
    const float* b_in  = (const float*)d_in[3];
    const float* W_ih  = (const float*)d_in[4];
    const float* W_hh  = (const float*)d_in[5];
    const float* b_ih  = (const float*)d_in[6];
    const float* b_hh  = (const float*)d_in[7];
    const float* W_out = (const float*)d_in[8];
    const float* b_out = (const float*)d_in[9];
    float* out = (float*)d_out;

    // ---- workspace carve-up (256-byte aligned) --------------------------
    char* ws = (char*)d_ws;
    size_t off = 0;
    auto alloc = [&](size_t bytes) -> char* {
        char* p = ws + off;
        off += (bytes + 255) & ~(size_t)255;
        return p;
    };
    unsigned short* x0   = (unsigned short*)alloc((size_t)NT * NG * HID * 2); // 187 MB
    unsigned short* hsB  = (unsigned short*)alloc((size_t)NT * NG * HID * 2); // 187 MB
    unsigned short* hb0  = (unsigned short*)alloc((size_t)NG * HID * 2);
    unsigned short* hb1  = (unsigned short*)alloc((size_t)NG * HID * 2);
    float*          cbuf = (float*)alloc((size_t)NG * HID * 4);
    unsigned short* wihF = (unsigned short*)alloc((size_t)G4 * HID * 2);
    unsigned short* whhF = (unsigned short*)alloc((size_t)G4 * HID * 2);
    float*          bias = (float*)alloc((size_t)G4 * 4);
    (void)ws_size; (void)in_sizes; (void)n_in; (void)out_size;

    // ---- prep -----------------------------------------------------------
    init_state<<<(NG * HID + 255) / 256, 256, 0, stream>>>(hb0, cbuf, b_ih, b_hh, bias);
    pack_w<<<(G4 * HID) / 256, 256, 0, stream>>>(W_ih, wihF);
    pack_w<<<(G4 * HID) / 256, 256, 0, stream>>>(W_hh, whhF);
    lin_in<<<NT * NG, HID, 0, stream>>>(x, W_in, b_in, x0);

    // ---- 365 sequential LSTM steps (graph-captured launches) ------------
    unsigned short* hbuf[2] = {hb0, hb1};
    for (int t = 0; t < NT; ++t) {
        const unsigned short* x0t = x0  + (size_t)t * NG * HID;
        unsigned short*       hst = hsB + (size_t)t * NG * HID;
        lstm_step<<<dim3(32, 4), 256, 0, stream>>>(
            x0t, wihF, whhF, bias, hbuf[t & 1], cbuf, hbuf[(t + 1) & 1], hst);
    }

    // ---- linearOut ------------------------------------------------------
    lin_out<<<(NT * NG + 7) / 8, 256, 0, stream>>>(hsB, W_out, b_out, out);
}
